// WindowAttention_27307402068324
// MI455X (gfx1250) — compile-verified, matrix operand order checked by reading, not run on hardware
//
#include <hip/hip_runtime.h>
#include <hip/hip_bf16.h>
#include <stdint.h>

// ---------------------------------------------------------------------------
// Types for CDNA5 WMMA (wave32): v_wmma_f32_16x16x32_bf16
// ---------------------------------------------------------------------------
typedef __attribute__((ext_vector_type(8)))  __bf16 v8bf;
typedef __attribute__((ext_vector_type(16))) __bf16 v16bf;
typedef __attribute__((ext_vector_type(8)))  float  v8f;
typedef __attribute__((ext_vector_type(4)))  float  vf4;
typedef __attribute__((ext_vector_type(4)))  int    v4i;

union FragBF { v16bf v; v8bf h[2]; };
union Chunk8 { v8bf v; __bf16 e[8]; };

#define M_TOT 100352        // 2048 windows * 49 tokens
#define K_DIM 512
#define LDS_STRIDE 40       // 32 bf16 + 8 pad -> conflict-free ds_load_b128
#define PV_STRIDE 72        // 64 bf16 + 8 pad
#define QSCALE 0.17677669529663687f   // 32^-0.5

// ---- CDNA5 async global->LDS copy (ASYNCcnt) with safe fallback -----------
#if defined(__has_builtin)
#  if __has_builtin(__builtin_amdgcn_global_load_async_to_lds_b128)
#    define HAVE_ASYNC 1
#  endif
#endif
#ifndef HAVE_ASYNC
#  define HAVE_ASYNC 0
#endif

typedef __attribute__((address_space(1))) v4i gv4i;   // global int4
typedef __attribute__((address_space(3))) v4i lv4i;   // LDS int4

__device__ inline void wait_async() {
#if HAVE_ASYNC
  asm volatile("s_wait_asynccnt 0x0" ::: "memory");
#endif
}

__device__ inline void async_cp16(const __bf16* g, __bf16* l) {
#if HAVE_ASYNC
  __builtin_amdgcn_global_load_async_to_lds_b128((gv4i*)g, (lv4i*)l, 0, 0);
#else
  *(v8bf*)l = *(const v8bf*)g;
#endif
}

__device__ inline v8f wmma_bf16(v16bf a, v16bf b, v8f c) {
  return __builtin_amdgcn_wmma_f32_16x16x32_bf16(
      false, a, false, b, (short)0, c, false, false);
}

// 16x32 bf16 fragment from LDS (A pattern; also matches B when the "row"
// dimension of the LDS array is the WMMA N dimension and columns are K).
__device__ inline v16bf ld_frag(const __bf16* base, int rowbase, int stride) {
  const int lane = threadIdx.x & 31;
  const int m  = rowbase + (lane & 15);
  const int hi = (lane >> 4) & 1;
  const __bf16* p = base + m * stride + hi * 8;
  FragBF f;
  f.h[0] = *(const v8bf*)(p);
  f.h[1] = *(const v8bf*)(p + 16);
  return f.v;
}

// ---------------------------------------------------------------------------
// fp32 -> bf16 bulk convert (x, w_qkv, w_proj), 16 elems/thread
// ---------------------------------------------------------------------------
__global__ __launch_bounds__(256)
void cvt_kernel(const float* __restrict__ in, __bf16* __restrict__ out, long long n)
{
  const long long i = ((long long)blockIdx.x * 256 + threadIdx.x) * 16;
  if (i + 16 <= n) {
    vf4 f0 = *(const vf4*)(in + i + 0);
    vf4 f1 = *(const vf4*)(in + i + 4);
    vf4 f2 = *(const vf4*)(in + i + 8);
    vf4 f3 = *(const vf4*)(in + i + 12);
    Chunk8 c0, c1;
#pragma unroll
    for (int k = 0; k < 4; ++k) {
      c0.e[k] = (__bf16)f0[k]; c0.e[4 + k] = (__bf16)f1[k];
      c1.e[k] = (__bf16)f2[k]; c1.e[4 + k] = (__bf16)f3[k];
    }
    *(v8bf*)(out + i)     = c0.v;
    *(v8bf*)(out + i + 8) = c1.v;
  }
}

// ---------------------------------------------------------------------------
// Fused bias+mask table: bm[w][h][row][col] = mask[w][row][col] + rb[h][rid]
// 64*16*49*49 floats = 9.8 MB (L2-resident for the attention pass).
// ---------------------------------------------------------------------------
#define BM_TOT (64 * 16 * 2401)
__global__ __launch_bounds__(256)
void biasmask_kernel(const float* __restrict__ mask, const float* __restrict__ rb,
                     float* __restrict__ bm)
{
  const int idx = blockIdx.x * 256 + threadIdx.x;
  if (idx >= BM_TOT) return;
  const int col = idx % 49;
  const int t1  = idx / 49;
  const int row = t1 % 49;
  const int t2  = t1 / 49;
  const int h   = t2 & 15;
  const int w   = t2 >> 4;
  const int qy = row / 7, qx = row - qy * 7;
  const int ky = col / 7, kx = col - ky * 7;
  const int rid = (ky - qy + 6) * 13 + (kx - qx + 6);   // key - query
  bm[idx] = mask[w * 2401 + row * 49 + col] + rb[h * 169 + rid];
}

// ---------------------------------------------------------------------------
// GEMM: C[M x N] = A[M x 512](bf16) * W[N x 512](bf16)^T + bias
//   MODE 0: scatter bf16 into qkv workspace [b][h][s(v,q,k)][tok49][d32];
//           Q (s==1) is pre-scaled by 32^-0.5.
//   MODE 1: write fp32 + bias to d_out [M x 512]
// 256 thr = 8 waves; WG tile 128(M) x 64(N); K-step 32; async double buffer.
// ---------------------------------------------------------------------------
template<int MODE>
__global__ __launch_bounds__(256)
void gemm_bf16_kernel(const __bf16* __restrict__ A,
                      const __bf16* __restrict__ W,
                      const float* __restrict__ bias,
                      void* __restrict__ Out)
{
  __shared__ __align__(16) __bf16 As[2][128 * LDS_STRIDE];
  __shared__ __align__(16) __bf16 Bs[2][64 * LDS_STRIDE];

  const int t  = threadIdx.x;
  const int n0 = blockIdx.x * 64;
  const int m0 = blockIdx.y * 128;

  const int arow = t >> 1, ahalf = t & 1;  // A: 128 rows x (2 x 16 bf16)
  const int brow = t >> 2, bq = t & 3;     // B: 64 rows x (4 x 8 bf16)

  const __bf16* agsrc = A + (size_t)(m0 + arow) * K_DIM + ahalf * 16;
  const __bf16* bgsrc = W + (size_t)(n0 + brow) * K_DIM + bq * 8;
  __bf16* aldst = &As[0][arow * LDS_STRIDE + ahalf * 16];
  __bf16* bldst = &Bs[0][brow * LDS_STRIDE + bq * 8];

  auto issue = [&](int kt) {
    const int k0  = kt * 32;
    const int off = (kt & 1) ? 128 * LDS_STRIDE : 0;   // As buffer offset
    const int bof = (kt & 1) ? 64 * LDS_STRIDE : 0;    // Bs buffer offset
    async_cp16(agsrc + k0,     aldst + off);
    async_cp16(agsrc + k0 + 8, aldst + off + 8);
    async_cp16(bgsrc + k0,     bldst + bof);
  };

  const int wave = t >> 5;
  const int wm = wave >> 1;   // 0..3
  const int wn = wave & 1;    // 0..1
  v8f acc[2][2] = {};

  issue(0);
  for (int kt = 0; kt < K_DIM / 32; ++kt) {
    wait_async();            // this wave's async writes to buf(kt) done
    __syncthreads();         // everyone's done -> buf(kt) valid, buf(kt^1) free
    if (kt + 1 < K_DIM / 32) issue(kt + 1);
    const __bf16* Ab = As[kt & 1];
    const __bf16* Bb = Bs[kt & 1];
    v16bf a0 = ld_frag(Ab, wm * 32,      LDS_STRIDE);
    v16bf a1 = ld_frag(Ab, wm * 32 + 16, LDS_STRIDE);
    v16bf b0 = ld_frag(Bb, wn * 32,      LDS_STRIDE);
    v16bf b1 = ld_frag(Bb, wn * 32 + 16, LDS_STRIDE);
    acc[0][0] = wmma_bf16(a0, b0, acc[0][0]);
    acc[0][1] = wmma_bf16(a0, b1, acc[0][1]);
    acc[1][0] = wmma_bf16(a1, b0, acc[1][0]);
    acc[1][1] = wmma_bf16(a1, b1, acc[1][1]);
  }

  // C layout: lane holds col n=(L&15); VGPR r holds row r + 8*(L>>4).
  const int lane = t & 31;
  const int nlo  = lane & 15;
  const int hi   = lane >> 4;
#pragma unroll
  for (int i = 0; i < 2; ++i) {
    const int rowbase = m0 + wm * 32 + i * 16 + 8 * hi;
    int bw[8], tk[8];
    if constexpr (MODE == 0) {
      const int bwin0 = rowbase / 49;
      const int tok0  = rowbase - bwin0 * 49;
#pragma unroll
      for (int r = 0; r < 8; ++r) {        // one div + incremental carry
        int tkv = tok0 + r, bwv = bwin0;
        if (tkv >= 49) { tkv -= 49; bwv += 1; }
        bw[r] = bwv; tk[r] = tkv;
      }
    }
#pragma unroll
    for (int j = 0; j < 2; ++j) {
      const int col = n0 + wn * 32 + j * 16 + nlo;
      const float bc = bias[col];
      if constexpr (MODE == 0) {
        const int s  = col >> 9;           // 0=v,1=q,2=k (reference order)
        const int hh = (col >> 5) & 15;
        const int dd = col & 31;
        const float qs = (s == 1) ? QSCALE : 1.0f;  // fold 1/sqrt(d) into Q
        __bf16* qout = (__bf16*)Out;
#pragma unroll
        for (int r = 0; r < 8; ++r) {
          const size_t idx = ((((size_t)bw[r] * 16 + hh) * 3 + s) * 49 + tk[r]) * 32 + dd;
          qout[idx] = (__bf16)((acc[i][j][r] + bc) * qs);
        }
      } else {
        float* out = (float*)Out;
#pragma unroll
        for (int r = 0; r < 8; ++r) {
          out[(size_t)(rowbase + r) * 512 + col] = acc[i][j][r] + bc;
        }
      }
    }
  }
}

// ---------------------------------------------------------------------------
// Attention: one wave per (window b, head h). N=49 padded to 64, d=32.
// S = Q K^T (Q pre-scaled) + bm-table; exact softmax; O = P V via LDS
// transpose; 1/rowsum folded into the O epilogue. Fully branchless gating.
// ---------------------------------------------------------------------------
#define AWPB 2
__global__ __launch_bounds__(32 * AWPB)
void attn_kernel(const __bf16* __restrict__ qkv,
                 const float* __restrict__ bm,
                 __bf16* __restrict__ aout)
{
  __shared__ __align__(16) __bf16 smem[AWPB][64 * LDS_STRIDE * 2 + 32 * PV_STRIDE + 16 * PV_STRIDE];
  const int wave = threadIdx.x >> 5;
  const int lane = threadIdx.x & 31;
  const int gid  = blockIdx.x * AWPB + wave;
  const int b = gid >> 4;
  const int h = gid & 15;

  __bf16* Qs = smem[wave];
  __bf16* Ks = Qs + 64 * LDS_STRIDE;
  __bf16* Vt = Ks + 64 * LDS_STRIDE;
  __bf16* Ps = Vt + 32 * PV_STRIDE;

  const size_t base = (size_t)(b * 16 + h) * 3 * 49 * 32;

  Chunk8 zero8;
#pragma unroll
  for (int i = 0; i < 8; ++i) zero8.e[i] = (__bf16)0.0f;

  // ---- Stage Q, K (row major) and V (transposed), zero-padding 49..63 ----
#pragma unroll
  for (int rr = 0; rr < 2; ++rr) {
    const int n = lane + rr * 32;
    const bool valid = (n < 49);
    const int ns = valid ? n : 0;
    const __bf16* pv = qkv + base + ((size_t)0 * 49 + ns) * 32;
    const __bf16* pq = qkv + base + ((size_t)1 * 49 + ns) * 32;
    const __bf16* pk = qkv + base + ((size_t)2 * 49 + ns) * 32;

    Chunk8 q0, q1, q2, q3, k0, k1, k2, k3, v0, v1, v2, v3;
    q0.v = valid ? *(const v8bf*)(pq + 0)  : zero8.v;
    q1.v = valid ? *(const v8bf*)(pq + 8)  : zero8.v;
    q2.v = valid ? *(const v8bf*)(pq + 16) : zero8.v;
    q3.v = valid ? *(const v8bf*)(pq + 24) : zero8.v;
    k0.v = valid ? *(const v8bf*)(pk + 0)  : zero8.v;
    k1.v = valid ? *(const v8bf*)(pk + 8)  : zero8.v;
    k2.v = valid ? *(const v8bf*)(pk + 16) : zero8.v;
    k3.v = valid ? *(const v8bf*)(pk + 24) : zero8.v;
    v0.v = valid ? *(const v8bf*)(pv + 0)  : zero8.v;
    v1.v = valid ? *(const v8bf*)(pv + 8)  : zero8.v;
    v2.v = valid ? *(const v8bf*)(pv + 16) : zero8.v;
    v3.v = valid ? *(const v8bf*)(pv + 24) : zero8.v;

    __bf16* qd = Qs + n * LDS_STRIDE;
    *(v8bf*)(qd + 0)  = q0.v; *(v8bf*)(qd + 8)  = q1.v;
    *(v8bf*)(qd + 16) = q2.v; *(v8bf*)(qd + 24) = q3.v;
    __bf16* kd = Ks + n * LDS_STRIDE;
    *(v8bf*)(kd + 0)  = k0.v; *(v8bf*)(kd + 8)  = k1.v;
    *(v8bf*)(kd + 16) = k2.v; *(v8bf*)(kd + 24) = k3.v;
#pragma unroll
    for (int i = 0; i < 8; ++i) {
      Vt[(0 + i)  * PV_STRIDE + n] = v0.e[i];
      Vt[(8 + i)  * PV_STRIDE + n] = v1.e[i];
      Vt[(16 + i) * PV_STRIDE + n] = v2.e[i];
      Vt[(24 + i) * PV_STRIDE + n] = v3.e[i];
    }
  }
  asm volatile("s_wait_dscnt 0" ::: "memory");

  const int nlo = lane & 15;
  const int hi  = lane >> 4;
  const float NEG = -1.0e30f;
  const float* bmw = bm + (size_t)(((b & 63) * 16) + h) * 2401;

  // per-lane key-column data (branchless, hoisted)
  int   colc[4];
  float colgate[4];
#pragma unroll
  for (int nt = 0; nt < 4; ++nt) {
    const int col = nt * 16 + nlo;
    const bool ok = (col < 49);
    colgate[nt] = ok ? 0.0f : NEG;
    colc[nt]    = ok ? col : 48;
  }

  for (int mt = 0; mt < 4; ++mt) {
    // ---- S tiles: 4 x (16x16), K = d = 32 ----
    v16bf aq = ld_frag(Qs, mt * 16, LDS_STRIDE);
    v8f s[4];
#pragma unroll
    for (int nt = 0; nt < 4; ++nt) {
      v16bf bk = ld_frag(Ks, nt * 16, LDS_STRIDE);
      v8f z = {};
      s[nt] = wmma_bf16(aq, bk, z);
    }

    // ---- + (bias+mask) table + additive gates; exact softmax per row ----
    float inv[8];
#pragma unroll
    for (int r = 0; r < 8; ++r) {
      const int row = mt * 16 + r + 8 * hi;
      const float rowgate = (row < 49) ? 0.0f : NEG;
      const int rowc = (row < 49) ? row : 48;
      const float* mrow = bmw + rowc * 49;
      float mx = -3.0e38f;
#pragma unroll
      for (int nt = 0; nt < 4; ++nt) {
        // one clamped (always in-bounds) load; OOB handled by -1e30 gates.
        const float v = s[nt][r] + mrow[colc[nt]] + rowgate + colgate[nt];
        s[nt][r] = v;
        mx = fmaxf(mx, v);
      }
      mx = fmaxf(mx, __shfl_xor(mx, 1, 32));
      mx = fmaxf(mx, __shfl_xor(mx, 2, 32));
      mx = fmaxf(mx, __shfl_xor(mx, 4, 32));
      mx = fmaxf(mx, __shfl_xor(mx, 8, 32));
      float sum = 0.f;
#pragma unroll
      for (int nt = 0; nt < 4; ++nt) {
        const float p = __expf(s[nt][r] - mx);
        s[nt][r] = p;
        sum += p;
      }
      sum += __shfl_xor(sum, 1, 32);
      sum += __shfl_xor(sum, 2, 32);
      sum += __shfl_xor(sum, 4, 32);
      sum += __shfl_xor(sum, 8, 32);
      inv[r] = 1.0f / sum;
    }

    // ---- transpose P through LDS (C layout -> A-fragment layout) ----
#pragma unroll
    for (int r = 0; r < 8; ++r)
#pragma unroll
      for (int nt = 0; nt < 4; ++nt)
        Ps[(r + 8 * hi) * PV_STRIDE + nt * 16 + nlo] = (__bf16)s[nt][r];
    asm volatile("s_wait_dscnt 0" ::: "memory");

    // ---- O(16x32) = P(16x64) * V(64x32); normalization in epilogue ----
    v16bf p0 = ld_frag(Ps,      0, PV_STRIDE);
    v16bf p1 = ld_frag(Ps + 32, 0, PV_STRIDE);
#pragma unroll
    for (int ct = 0; ct < 2; ++ct) {
      v16bf vb0 = ld_frag(Vt,      ct * 16, PV_STRIDE);
      v16bf vb1 = ld_frag(Vt + 32, ct * 16, PV_STRIDE);
      v8f z = {};
      v8f o = wmma_bf16(p0, vb0, z);
      o = wmma_bf16(p1, vb1, o);
#pragma unroll
      for (int r = 0; r < 8; ++r) {
        const int row = mt * 16 + r + 8 * hi;
        if (row < 49) {
          aout[((size_t)b * 49 + row) * 512 + h * 32 + ct * 16 + nlo] =
              (__bf16)(o[r] * inv[r]);
        }
      }
    }
  }
}

// ---------------------------------------------------------------------------
// Launch: cvt x/weights -> bias+mask table -> QKV gemm -> attention -> proj
// ---------------------------------------------------------------------------
extern "C" void kernel_launch(void* const* d_in, const int* in_sizes, int n_in,
                              void* d_out, int out_size, void* d_ws, size_t ws_size,
                              hipStream_t stream)
{
  (void)in_sizes; (void)n_in; (void)out_size; (void)ws_size;
  const float* x      = (const float*)d_in[0];
  const float* mask   = (const float*)d_in[1];
  const float* w_qkv  = (const float*)d_in[2];
  const float* b_qkv  = (const float*)d_in[3];
  const float* w_proj = (const float*)d_in[4];
  const float* b_proj = (const float*)d_in[5];
  const float* rb     = (const float*)d_in[6];
  float* out = (float*)d_out;

  const long long XN  = (long long)M_TOT * K_DIM;        // 51,380,224
  const long long WQN = (long long)1536 * 512;           //    786,432
  const long long WPN = (long long)512 * 512;            //    262,144
  const size_t QKV_ELEMS  = (size_t)2048 * 16 * 3 * 49 * 32;
  const size_t ATTN_ELEMS = (size_t)M_TOT * 512;

  __bf16* x_bf     = (__bf16*)d_ws;
  __bf16* wqkv_bf  = x_bf + XN;
  __bf16* wproj_bf = wqkv_bf + WQN;
  __bf16* qkv_ws   = wproj_bf + WPN;                     // [b][h][vqk][49][32]
  __bf16* attn_ws  = qkv_ws + QKV_ELEMS;                 // [100352][512]
  float*  bm_ws    = (float*)(attn_ws + ATTN_ELEMS);     // [64][16][49][49]

  cvt_kernel<<<dim3((unsigned)(XN / (16 * 256))),  dim3(256), 0, stream>>>(x, x_bf, XN);
  cvt_kernel<<<dim3((unsigned)(WQN / (16 * 256))), dim3(256), 0, stream>>>(w_qkv, wqkv_bf, WQN);
  cvt_kernel<<<dim3((unsigned)(WPN / (16 * 256))), dim3(256), 0, stream>>>(w_proj, wproj_bf, WPN);
  biasmask_kernel<<<dim3((BM_TOT + 255) / 256), dim3(256), 0, stream>>>(mask, rb, bm_ws);

  dim3 b256(256);
  dim3 g_qkv(1536 / 64, M_TOT / 128);    // x = N tiles (L2-hot weights)
  gemm_bf16_kernel<0><<<g_qkv, b256, 0, stream>>>(x_bf, wqkv_bf, b_qkv, qkv_ws);

  dim3 g_attn((2048 * 16) / AWPB);
  attn_kernel<<<g_attn, dim3(32 * AWPB), 0, stream>>>(qkv_ws, bm_ws, attn_ws);

  dim3 g_proj(512 / 64, M_TOT / 128);
  gemm_bf16_kernel<1><<<g_proj, b256, 0, stream>>>(attn_ws, wproj_bf, b_proj, out);
}
